// FNO3d_21990232555558
// MI455X (gfx1250) — matrix operand order repaired and checked by reading, not compiled, not run on hardware
//
#include <hip/hip_runtime.h>
#include <hip/hip_bf16.h>

// ---------------------------------------------------------------------------
// FNO2d forward for MI455X (gfx1250, wave32, WMMA bf16, f32 accumulate).
//   k_wpack  : per-layer repack of spectral weights -> bf16 [site][o][i]
//   k_fwdx   : partial rfft along W (20 modes), LDS rows + twiddles + prefetch
//   k_fwdy   : partial fft along H (40 ky modes)
//   k_mix    : per-mode 64x64 complex channel mix, packed-B WMMA (no spills)
//   k_skip   : pointwise 64x64 skip GEMM, 8 WMMA/wave, b128 fragment loads
//   k_invx   : inverse partial DFT along W -> interleaved float2 G
//   k_invy   : inverse partial DFT along H as GEMM (K=80 pad 96), immediate-
//              offset G loads, coalesced stores into spec[y][bo][x]
//   k_combine: LDS-tiled transpose + skip add + GELU
//   k_fc1    : 64->128 GEMM (16 WMMA/wave) + GELU;  k_fc2: 128->2 VALU
// ---------------------------------------------------------------------------

typedef __attribute__((ext_vector_type(16))) __bf16 v16bf;
typedef __attribute__((ext_vector_type(8)))  __bf16 bf16x8;
typedef __attribute__((ext_vector_type(8)))  float  v8f;

#define WMMA_BF16(A, B, C) \
  __builtin_amdgcn_wmma_f32_16x16x32_bf16(false, (A), false, (B), (short)0, (C), false, false)

#define TWO_PI_256 0.0245436926061702596f  // 2*pi/256

static constexpr int Bn = 16, Hn = 256, Wn = 256, Cw = 64;
static constexpr int M1n = 20, M2n = 20, NL = 4, PR = 128;
static constexpr long long Pn = (long long)Bn * Hn * Wn;   // 1,048,576 pixels
static constexpr long long NSPEC = 1024LL * 256;           // bo*x per y-row
static constexpr int NSITE = 40 * M2n;                     // 800 mode sites

__device__ __forceinline__ void fill8(v16bf& F, int base, float4 a, float4 b) {
  F[base + 0] = (__bf16)a.x; F[base + 1] = (__bf16)a.y;
  F[base + 2] = (__bf16)a.z; F[base + 3] = (__bf16)a.w;
  F[base + 4] = (__bf16)b.x; F[base + 5] = (__bf16)b.y;
  F[base + 6] = (__bf16)b.z; F[base + 7] = (__bf16)b.w;
}

// A-fragment from f32 row (16x32): lane row M=lane&15; K runs [kb..kb+15],
// kb = 16*(lane>=16). Four b128 loads, immediate offsets.
__device__ __forceinline__ void load_afrag_f32(v16bf& A, const float* rowp, int lane) {
  int kb = lane & 16;
  float4 a0 = *(const float4*)(rowp + kb);
  float4 a1 = *(const float4*)(rowp + kb + 4);
  float4 a2 = *(const float4*)(rowp + kb + 8);
  float4 a3 = *(const float4*)(rowp + kb + 12);
  fill8(A, 0, a0, a1);
  fill8(A, 8, a2, a3);
}

// B-fragment from f32 column K-run (32x16): lane col N=lane&15; K = e + 16*(lane>=16).
__device__ __forceinline__ void load_bfrag_f32(v16bf& B, const float* colp, int lane) {
  int kb = lane & 16;
  float4 b0 = *(const float4*)(colp + kb);
  float4 b1 = *(const float4*)(colp + kb + 4);
  float4 b2 = *(const float4*)(colp + kb + 8);
  float4 b3 = *(const float4*)(colp + kb + 12);
  fill8(B, 0, b0, b1);
  fill8(B, 8, b2, b3);
}

// B-fragment from pre-packed bf16 column K-run: two 16-byte loads.
__device__ __forceinline__ void load_bfrag_bf16(v16bf& B, const __bf16* colp, int lane) {
  int kb = lane & 16;
  bf16x8 b0 = *(const bf16x8*)(colp + kb);
  bf16x8 b1 = *(const bf16x8*)(colp + kb + 8);
#pragma unroll
  for (int e = 0; e < 8; ++e) { B[e] = b0[e]; B[8 + e] = b1[e]; }
}

__device__ __forceinline__ float gelu_tanh(float x) {
  const float k0 = 0.7978845608028654f, k1 = 0.044715f;
  return 0.5f * x * (1.0f + tanhf(k0 * (x + k1 * x * x * x)));
}

// ---------------------------------------------------------------------------
// Lifting: h[b,y,x,c] = x[pix]*fc0_w[c] + fc0_b[c]   (channels-last, float4)
// ---------------------------------------------------------------------------
__global__ void k_lift(const float* __restrict__ x, const float4* __restrict__ w4,
                       const float4* __restrict__ b4, float4* __restrict__ h4,
                       long long total) {
  long long i = (long long)blockIdx.x * blockDim.x + threadIdx.x;
  if (i >= total) return;
  int cq = (int)(i & 15);
  long long pix = i >> 4;
  float xv = x[pix];
  float4 wv = w4[cq], bv = b4[cq];
  h4[i] = make_float4(xv * wv.x + bv.x, xv * wv.y + bv.y,
                      xv * wv.z + bv.z, xv * wv.w + bv.w);
}

// ---------------------------------------------------------------------------
// Repack one layer's spectral weights into bf16 [site][o][i] (K contiguous).
// ---------------------------------------------------------------------------
__global__ void k_wpack(const float* __restrict__ w1r, const float* __restrict__ w1i,
                        const float* __restrict__ w2r, const float* __restrict__ w2i,
                        __bf16* __restrict__ pr, __bf16* __restrict__ pi) {
  long long i = (long long)blockIdx.x * blockDim.x + threadIdx.x;
  if (i >= (long long)NSITE * 64 * 64) return;
  int ii = (int)(i & 63);            // input channel (K)
  int oo = (int)((i >> 6) & 63);     // output channel (N)
  int site = (int)(i >> 12);         // kyI*20 + kx
  int kx = site % M2n, kyI = site / M2n;
  const float* sr;
  const float* si;
  int m1;
  if (kyI < M1n) { sr = w1r; si = w1i; m1 = kyI; }
  else           { sr = w2r; si = w2i; m1 = kyI - M1n; }
  long long widx = ((long long)ii * 64 + oo) * 400 + m1 * M2n + kx;
  pr[i] = (__bf16)sr[widx];
  pi[i] = (__bf16)si[widx];
}

// ---------------------------------------------------------------------------
// Forward partial DFT along W: Fx[b,y,kx,c], kx in [0,20)
// ---------------------------------------------------------------------------
__global__ void k_fwdx(const float* __restrict__ h, float* __restrict__ fxr,
                       float* __restrict__ fxi) {
  __shared__ float ct[256], st[256];
  __shared__ float srow[256 * 16];  // [x][ci]
  int tid = threadIdx.x;
  {
    float ang = TWO_PI_256 * (float)tid;
    ct[tid] = __cosf(ang);
    st[tid] = __sinf(ang);
  }
  int by = blockIdx.x;  // b*256 + y
  const float* row = h + (long long)by * Wn * Cw;
  for (int cc = 0; cc < 4; ++cc) {
    __syncthreads();
    for (int t = tid; t < 256 * 16; t += 256) {
      int xx = t >> 4, ci = t & 15;
      srow[xx * 16 + ci] = row[(long long)xx * Cw + cc * 16 + ci];
    }
    __syncthreads();
    if (cc < 3)  // prefetch next channel chunk (global_prefetch_b8)
      __builtin_prefetch(row + (long long)tid * Cw + (cc + 1) * 16, 0, 1);
    for (int o = tid; o < M2n * 16; o += 256) {
      int kx = o >> 4, ci = o & 15;
      float re = 0.f, im = 0.f;
      for (int xx = 0; xx < 256; ++xx) {
        float v = srow[xx * 16 + ci];
        int t = (kx * xx) & 255;
        re += v * ct[t];
        im -= v * st[t];
      }
      long long oidx = ((long long)by * M2n + kx) * Cw + (cc * 16 + ci);
      fxr[oidx] = re;
      fxi[oidx] = im;
    }
  }
}

// ---------------------------------------------------------------------------
// Forward partial DFT along H (40 ky modes) -> Xf[(site)*16+b][c]
// ---------------------------------------------------------------------------
__global__ void k_fwdy(const float* __restrict__ fxr, const float* __restrict__ fxi,
                       float* __restrict__ xfr, float* __restrict__ xfi) {
  __shared__ float ct[256], st[256];
  int tid = threadIdx.x;
  {
    float ang = TWO_PI_256 * (float)tid;
    ct[tid] = __cosf(ang);
    st[tid] = __sinf(ang);
  }
  __syncthreads();
  long long i = (long long)blockIdx.x * blockDim.x + tid;
  if (i >= (long long)Bn * 40 * M2n * Cw) return;
  int c = (int)(i & 63);
  long long r = i >> 6;
  int kx = (int)(r % M2n);
  r /= M2n;
  int kyI = (int)(r % 40);
  int b = (int)(r / 40);
  int ky = (kyI < M1n) ? kyI : (216 + kyI);  // 236..255
  float re = 0.f, im = 0.f;
  for (int y = 0; y < 256; ++y) {
    long long idx = (((long long)(b * 256 + y)) * M2n + kx) * Cw + c;
    float xr = fxr[idx], xi = fxi[idx];
    int t = (ky * y) & 255;
    float cc = ct[t], ss = st[t];
    re += xr * cc + xi * ss;
    im += xi * cc - xr * ss;
  }
  long long o = (((long long)(kyI * M2n + kx)) * 16 + b) * Cw + c;
  xfr[o] = re;
  xfi[o] = im;
}

// ---------------------------------------------------------------------------
// Mode mix: one wave per (site, 2 o-tiles); packed bf16 B; 16 WMMA/wave.
// ---------------------------------------------------------------------------
__global__ void k_mix(const float* __restrict__ xfr, const float* __restrict__ xfi,
                      const __bf16* __restrict__ pr, const __bf16* __restrict__ pi,
                      float* __restrict__ yfr, float* __restrict__ yfi) {
  int lane = threadIdx.x & 31;
  int wave = (int)(((long long)blockIdx.x * blockDim.x + threadIdx.x) >> 5);
  if (wave >= NSITE * 2) return;
  int site = wave >> 1;
  int half = wave & 1;     // o-tiles {2h, 2h+1}
  int kx = site % M2n, kyI = site / M2n;
  long long abase = (long long)site * 16 * Cw;
  long long wbase = (long long)site * 64 * 64;
  int row = lane & 15, col = lane & 15;

  v8f z = {};
  v8f accr[2] = {z, z};
  v8f acci[2] = {z, z};
#pragma unroll
  for (int kc = 0; kc < 2; ++kc) {
    int k0 = kc * 32;
    v16bf Ar, Ai, AiN;
    load_afrag_f32(Ar, xfr + abase + (long long)row * Cw + k0, lane);
    {
      int kb = lane & 16;
      const float* rp = xfi + abase + (long long)row * Cw + k0 + kb;
      float4 a0 = *(const float4*)(rp);
      float4 a1 = *(const float4*)(rp + 4);
      float4 a2 = *(const float4*)(rp + 8);
      float4 a3 = *(const float4*)(rp + 12);
      fill8(Ai, 0, a0, a1);
      fill8(Ai, 8, a2, a3);
      a0.x = -a0.x; a0.y = -a0.y; a0.z = -a0.z; a0.w = -a0.w;
      a1.x = -a1.x; a1.y = -a1.y; a1.z = -a1.z; a1.w = -a1.w;
      a2.x = -a2.x; a2.y = -a2.y; a2.z = -a2.z; a2.w = -a2.w;
      a3.x = -a3.x; a3.y = -a3.y; a3.z = -a3.z; a3.w = -a3.w;
      fill8(AiN, 0, a0, a1);
      fill8(AiN, 8, a2, a3);
    }
#pragma unroll
    for (int t = 0; t < 2; ++t) {
      int nt = half * 2 + t;
      const __bf16* cb = pr + wbase + (long long)(nt * 16 + col) * 64 + k0;
      const __bf16* ci = pi + wbase + (long long)(nt * 16 + col) * 64 + k0;
      v16bf Br, Bi;
      load_bfrag_bf16(Br, cb, lane);
      load_bfrag_bf16(Bi, ci, lane);
      accr[t] = WMMA_BF16(Ar, Br, accr[t]);
      accr[t] = WMMA_BF16(AiN, Bi, accr[t]);  // re -= Aim*Bim
      acci[t] = WMMA_BF16(Ar, Bi, acci[t]);
      acci[t] = WMMA_BF16(Ai, Br, acci[t]);
    }
  }
#pragma unroll
  for (int t = 0; t < 2; ++t) {
    int oo = (half * 2 + t) * 16 + col;
#pragma unroll
    for (int r2 = 0; r2 < 8; ++r2) {
      int bb = r2 + ((lane & 16) ? 8 : 0);
      long long oidx = (((long long)bb * Cw + oo) * 40 + kyI) * M2n + kx;
      yfr[oidx] = accr[t][r2];
      yfi[oidx] = acci[t][r2];
    }
  }
}

// ---------------------------------------------------------------------------
// Skip GEMM: one wave per 16-row tile, all 4 o-tiles; 8 WMMA/wave.
// ---------------------------------------------------------------------------
__global__ void k_skip(const float* __restrict__ h, const float* __restrict__ sw,
                       const float* __restrict__ sb, float* __restrict__ h2) {
  int lane = threadIdx.x & 31;
  long long wave = ((long long)blockIdx.x * blockDim.x + threadIdx.x) >> 5;
  long long mbase = wave * 16;
  int row = lane & 15, col = lane & 15;
  v8f z = {};
  v8f acc[4] = {z, z, z, z};
#pragma unroll
  for (int kc = 0; kc < 2; ++kc) {
    int k0 = kc * 32;
    v16bf A;
    load_afrag_f32(A, h + (mbase + row) * Cw + k0, lane);
#pragma unroll
    for (int nt = 0; nt < 4; ++nt) {
      v16bf Bw;
      load_bfrag_f32(Bw, sw + (nt * 16 + col) * Cw + k0, lane);
      acc[nt] = WMMA_BF16(A, Bw, acc[nt]);
    }
  }
#pragma unroll
  for (int nt = 0; nt < 4; ++nt) {
    float bias = sb[nt * 16 + col];
#pragma unroll
    for (int r = 0; r < 8; ++r) {
      long long m = mbase + r + ((lane & 16) ? 8 : 0);
      h2[m * Cw + nt * 16 + col] = acc[nt][r] + bias;
    }
  }
}

// ---------------------------------------------------------------------------
// Inverse partial DFT along W -> interleaved complex G (float2), incl scale.
// ---------------------------------------------------------------------------
__global__ void k_invx(const float* __restrict__ yfr, const float* __restrict__ yfi,
                       float2* __restrict__ g2) {
  __shared__ float ct[256], st[256];
  __shared__ float yr[M2n], yi[M2n];
  int tid = threadIdx.x;
  {
    float ang = TWO_PI_256 * (float)tid;
    ct[tid] = __cosf(ang);
    st[tid] = __sinf(ang);
  }
  long long blk = blockIdx.x;  // (b*64+o)*40 + kyI
  if (tid < M2n) {
    yr[tid] = yfr[blk * M2n + tid];
    yi[tid] = yfi[blk * M2n + tid];
  }
  __syncthreads();
  int x = tid;
  float re = 0.f, im = 0.f;
#pragma unroll
  for (int kx = 0; kx < M2n; ++kx) {
    float wgt = (kx == 0) ? 1.0f : 2.0f;
    int t = (kx * x) & 255;
    float cc = ct[t], ss = st[t];
    re += wgt * (yr[kx] * cc - yi[kx] * ss);
    im += wgt * (yr[kx] * ss + yi[kx] * cc);
  }
  const float scl = 1.0f / 65536.0f;  // 1/(H*W)
  g2[blk * 256 + x] = make_float2(re * scl, im * scl);
}

// Precompute inverse-y DFT matrix Ay[256 y][96 k] bf16 (K=80 real, zero pad)
__global__ void k_ayinit(__bf16* __restrict__ ay) {
  int i = blockIdx.x * blockDim.x + threadIdx.x;
  if (i >= 256 * 96) return;
  int k = i % 96, y = i / 96;
  float v = 0.f;
  if (k < 80) {
    int kyI = k >> 1;
    int ky = (kyI < M1n) ? kyI : (216 + kyI);
    float ang = TWO_PI_256 * (float)((ky * y) & 255);
    v = ((k & 1) == 0) ? __cosf(ang) : -__sinf(ang);
  }
  ay[i] = (__bf16)v;
}

// ---------------------------------------------------------------------------
// Inverse partial DFT along H as GEMM: spec[y][bo][x] = sum_k Ay[y,k]*G[k,bo,x]
// G loads: one lane-base pointer + constant immediate offsets (j*256 float2).
// ---------------------------------------------------------------------------
__global__ void k_invy(const __bf16* __restrict__ ay, const float2* __restrict__ g2,
                       float* __restrict__ spec) {
  int lane = threadIdx.x & 31;
  long long wave = ((long long)blockIdx.x * blockDim.x + threadIdx.x) >> 5;
  int mt = (int)(wave & 15);
  long long ntile = wave >> 4;  // 0..16383
  int ybase = mt * 16;
  long long nbase = ntile * 16;
  int bo = (int)(nbase >> 8);
  int xbase = (int)(nbase & 255);
  int row = lane & 15, col = lane & 15;
  int kbA = lane & 16;
  int kbB = lane & 16;
  int xcol = xbase + col;
  v8f acc = {};
#pragma unroll
  for (int kc = 0; kc < 3; ++kc) {
    int k0 = kc * 32;
    v16bf A;
    {
      const __bf16* ar = ay + (ybase + row) * 96 + k0 + kbA;
      bf16x8 a0 = *(const bf16x8*)(ar);
      bf16x8 a1 = *(const bf16x8*)(ar + 8);
#pragma unroll
      for (int e = 0; e < 8; ++e) { A[e] = a0[e]; A[8 + e] = a1[e]; }
    }
    v16bf Bv;
    int jb = (k0 + kbB) >> 1;  // kyI base for this lane half
    const float2* gp = g2 + ((long long)bo * 40 + jb) * 256 + xcol;
#pragma unroll
    for (int j = 0; j < 8; ++j) {
      int kyI = jb + j;
      float2 v = make_float2(0.f, 0.f);
      if (kyI < 40) v = gp[j * 256];  // immediate offset
      Bv[2 * j] = (__bf16)v.x;
      Bv[2 * j + 1] = (__bf16)v.y;
    }
    acc = WMMA_BF16(A, Bv, acc);
  }
#pragma unroll
  for (int r = 0; r < 8; ++r) {
    int y = ybase + r + ((lane & 16) ? 8 : 0);
    spec[(long long)y * NSPEC + nbase + col] = acc[r];
  }
}

// ---------------------------------------------------------------------------
// Combine: h[b,y,x,o] = act(spec[y][b*64+o][x] + h2[pix,o]); LDS transpose.
// ---------------------------------------------------------------------------
__global__ void k_combine(const float* __restrict__ spec, const float* __restrict__ h2,
                          float* __restrict__ h, int apply_gelu) {
  __shared__ float tile[64][65];
  int tid = threadIdx.x;
  int blk = blockIdx.x;  // (b*256+y)*4 + xc
  int xc = blk & 3;
  int by = blk >> 2;
  int b = by >> 8;
  int y = by & 255;
  int x0 = xc * 64;
  const float* sp = spec + (long long)y * NSPEC + (long long)b * 64 * 256 + x0;
  for (int i = tid; i < 64 * 64; i += 256) {
    int xx = i & 63, oo = i >> 6;                 // coalesced along x
    tile[oo][xx] = sp[(long long)oo * 256 + xx];
  }
  __syncthreads();
  long long pixbase = (long long)by * 256 + x0;
  for (int i = tid; i < 64 * 64; i += 256) {
    int oo = i & 63, xx = i >> 6;                 // coalesced along o
    long long idx = (pixbase + xx) * Cw + oo;
    float v = tile[oo][xx] + h2[idx];
    if (apply_gelu) v = gelu_tanh(v);
    h[idx] = v;
  }
}

// ---------------------------------------------------------------------------
// fc1: one wave per 16-row tile, all 8 o-tiles; 16 WMMA/wave; GELU.
// ---------------------------------------------------------------------------
__global__ void k_fc1(const float* __restrict__ h, const float* __restrict__ w,
                      const float* __restrict__ bias, float* __restrict__ hp) {
  int lane = threadIdx.x & 31;
  long long wave = ((long long)blockIdx.x * blockDim.x + threadIdx.x) >> 5;
  long long mbase = wave * 16;
  int row = lane & 15, col = lane & 15;
  v8f z = {};
  v8f acc[8] = {z, z, z, z, z, z, z, z};
#pragma unroll
  for (int kc = 0; kc < 2; ++kc) {
    int k0 = kc * 32;
    v16bf A;
    load_afrag_f32(A, h + (mbase + row) * Cw + k0, lane);
#pragma unroll
    for (int nt = 0; nt < 8; ++nt) {
      v16bf Bw;
      load_bfrag_f32(Bw, w + (nt * 16 + col) * Cw + k0, lane);
      acc[nt] = WMMA_BF16(A, Bw, acc[nt]);
    }
  }
#pragma unroll
  for (int nt = 0; nt < 8; ++nt) {
    float bv = bias[nt * 16 + col];
#pragma unroll
    for (int r = 0; r < 8; ++r) {
      long long m = mbase + r + ((lane & 16) ? 8 : 0);
      hp[m * PR + nt * 16 + col] = gelu_tanh(acc[nt][r] + bv);
    }
  }
}

// ---------------------------------------------------------------------------
// fc2: out[b,{0,1},y,x], vectorized float4 dot over 128 channels.
// ---------------------------------------------------------------------------
__global__ void k_fc2(const float* __restrict__ hp, const float* __restrict__ w,
                      const float* __restrict__ bias, float* __restrict__ out) {
  long long pix = (long long)blockIdx.x * blockDim.x + threadIdx.x;
  if (pix >= Pn) return;
  const float4* hv = (const float4*)(hp + pix * PR);
  const float4* w0 = (const float4*)(w);
  const float4* w1 = (const float4*)(w + PR);
  float a0 = bias[0], a1 = bias[1];
#pragma unroll 8
  for (int i = 0; i < PR / 4; ++i) {
    float4 v = hv[i], p = w0[i], q = w1[i];
    a0 += v.x * p.x + v.y * p.y + v.z * p.z + v.w * p.w;
    a1 += v.x * q.x + v.y * q.y + v.z * q.z + v.w * q.w;
  }
  long long bIdx = pix >> 16;
  long long yx = pix & 65535;
  out[(bIdx * 2 + 0) * 65536 + yx] = a0;
  out[(bIdx * 2 + 1) * 65536 + yx] = a1;
}

// ---------------------------------------------------------------------------
extern "C" void kernel_launch(void* const* d_in, const int* in_sizes, int n_in,
                              void* d_out, int out_size, void* d_ws, size_t ws_size,
                              hipStream_t stream) {
  (void)in_sizes; (void)n_in; (void)out_size; (void)ws_size;
  const float* x      = (const float*)d_in[0];
  const float* fc0_w  = (const float*)d_in[1];
  const float* fc0_b  = (const float*)d_in[2];
  const float* w1r    = (const float*)d_in[3];
  const float* w1i    = (const float*)d_in[4];
  const float* w2r    = (const float*)d_in[5];
  const float* w2i    = (const float*)d_in[6];
  const float* skw    = (const float*)d_in[7];
  const float* skb    = (const float*)d_in[8];
  const float* fc1_w  = (const float*)d_in[9];
  const float* fc1_b  = (const float*)d_in[10];
  const float* fc2_w  = (const float*)d_in[11];
  const float* fc2_b  = (const float*)d_in[12];
  float* out = (float*)d_out;

  // ---- workspace carve (floats) ----
  float* ws = (float*)d_ws;
  long long off = 0;
  float* h    = ws + off; off += Pn * 64;    // activation, channels-last
  float* big  = ws + off; off += Pn * 128;   // [h2 | spec] during layers; hp after
  float* h2   = big;                         // Pn*64
  float* spec = big + Pn * 64;               // 256 * NSPEC = Pn*64
  float* hp   = big;                         // Pn*128 (after layers)
  float* fxr = ws + off; off += (long long)Bn * Hn * M2n * Cw;
  float* fxi = ws + off; off += (long long)Bn * Hn * M2n * Cw;
  float* xfr = ws + off; off += (long long)NSITE * 16 * Cw;
  float* xfi = ws + off; off += (long long)NSITE * 16 * Cw;
  float* yfr = ws + off; off += (long long)Bn * Cw * 40 * M2n;
  float* yfi = ws + off; off += (long long)Bn * Cw * 40 * M2n;
  float2* g2 = (float2*)(ws + off); off += (long long)Bn * Cw * 40 * 256 * 2;
  __bf16* ay = (__bf16*)(ws + off); off += (256 * 96 + 1) / 2;
  __bf16* wpr = (__bf16*)(ws + off); off += (long long)NSITE * 64 * 64 / 2;
  __bf16* wpi = (__bf16*)(ws + off); off += (long long)NSITE * 64 * 64 / 2;

  const long long specWstride = 64LL * 64 * 400;  // per-layer spectral weights

  k_ayinit<<<dim3(96), dim3(256), 0, stream>>>(ay);

  k_lift<<<dim3((unsigned)((Pn * 16 + 255) / 256)), dim3(256), 0, stream>>>(
      x, (const float4*)fc0_w, (const float4*)fc0_b, (float4*)h, Pn * 16);

  for (int l = 0; l < NL; ++l) {
    // repack spectral weights (800*4096 elems -> 12800 blocks)
    k_wpack<<<dim3((unsigned)(((long long)NSITE * 4096 + 255) / 256)), dim3(256), 0, stream>>>(
        w1r + l * specWstride, w1i + l * specWstride,
        w2r + l * specWstride, w2i + l * specWstride, wpr, wpi);
    k_fwdx<<<dim3(Bn * Hn), dim3(256), 0, stream>>>(h, fxr, fxi);
    k_fwdy<<<dim3((unsigned)((Bn * 40LL * M2n * Cw + 255) / 256)), dim3(256), 0, stream>>>(
        fxr, fxi, xfr, xfi);
    // mix: 1600 waves -> 200 blocks
    k_mix<<<dim3(200), dim3(256), 0, stream>>>(xfr, xfi, wpr, wpi, yfr, yfi);
    // skip: waves = Pn/16 = 65536 -> 8192 blocks
    k_skip<<<dim3(8192), dim3(256), 0, stream>>>(h, skw + l * 64 * 64, skb + l * 64, h2);
    k_invx<<<dim3(Bn * Cw * 40), dim3(256), 0, stream>>>(yfr, yfi, g2);
    // invy: waves = 16 * 16384 -> 32768 blocks
    k_invy<<<dim3(32768), dim3(256), 0, stream>>>(ay, g2, spec);
    k_combine<<<dim3(Bn * Hn * 4), dim3(256), 0, stream>>>(
        spec, h2, h, (l < NL - 1) ? 1 : 0);
  }

  // fc1: waves = Pn/16 = 65536 -> 8192 blocks
  k_fc1<<<dim3(8192), dim3(256), 0, stream>>>(h, fc1_w, fc1_b, hp);
  k_fc2<<<dim3((unsigned)((Pn + 255) / 256)), dim3(256), 0, stream>>>(hp, fc2_w, fc2_b, out);
}
